// SKVMN_84146999263911
// MI455X (gfx1250) — matrix operand order repaired and checked by reading, not compiled, not run on hardware
//
#include <hip/hip_runtime.h>
#include <hip/hip_bf16.h>

typedef __attribute__((ext_vector_type(16))) __bf16 v16bf;
typedef __attribute__((ext_vector_type(8)))  float  v8f;

#define NUMC 4096
#define BSZ  32
#define LT   512
#define DD   128
#define MM   50
#define MPAD 64
#define BPB  8     // batch rows handled per persistent block (within one 16-row WMMA tile)

// ---------------- workspace layout (bytes) ----------------
static constexpr size_t WS_KBF  = 0;                                    // B*L*D bf16
static constexpr size_t WS_YQ   = WS_KBF  + (size_t)BSZ*LT*DD*2;        // B*L*D f32
static constexpr size_t WS_FBF  = WS_YQ   + (size_t)BSZ*LT*DD*4;        // B*L*D bf16 (f_all)
static constexpr size_t WS_WALL = WS_FBF  + (size_t)BSZ*LT*DD*2;        // B*L*M f32
static constexpr size_t WS_CODE = WS_WALL + (size_t)BSZ*LT*MM*4;        // B*L*2 u64
static constexpr size_t WS_PREV = WS_CODE + (size_t)BSZ*LT*16;          // B*L i32
static constexpr size_t WS_HH   = WS_PREV + (size_t)BSZ*LT*4;           // L*B*D f32
static constexpr size_t WS_CH   = WS_HH   + (size_t)LT*BSZ*DD*4;        // L*B*D f32
static constexpr size_t WS_WTAB = WS_CH   + (size_t)LT*BSZ*DD*4;        // bf16 weight table

// bf16 weight table element offsets (natural (out,in) layout -> WMMA B-matrix ready)
#define WOFF_MK   0        /* MkPad 64x128  */
#define WOFF_FW   8192     /* f_W  128x256  */
#define WOFF_AWF  40960    /* a_Wf 128x128  */
#define WOFF_EW   57344    /* e_W  128x128  */
#define WOFF_ADW  73728    /* add_W 128x128 */
#define WOFF_WIH  90112    /* Wih 512x128   */
#define WOFF_WHH  155648   /* Whh 512x128   */
#define WT_TOTAL  221184

__device__ __forceinline__ unsigned short f2bf(float x) {
  union { float f; unsigned int u; } v; v.f = x;
  unsigned int r = v.u + 0x7FFFu + ((v.u >> 16) & 1u);   // round-to-nearest-even
  return (unsigned short)(r >> 16);
}
__device__ __forceinline__ float bf2f(unsigned short h) {
  union { unsigned int u; float f; } v; v.u = ((unsigned int)h) << 16; return v.f;
}
__device__ __forceinline__ float sigmf(float x) { return 1.0f / (1.0f + __expf(-x)); }

union FragU { v16bf v; unsigned int u[8]; };

// A-matrix fragment (16x32 bf16 tile): act is row-major [row][K], stride in elements.
// ISA 7.12.2: lane l -> row l&15; VGPR i -> K = 2*(i&3) + 16*(i>>2) + 8*(l>>4).
__device__ __forceinline__ v16bf fragA(const unsigned short* act, int stride, int k0, int lane) {
  FragU f;
  const unsigned short* p = act + (size_t)(lane & 15) * stride + k0 + ((lane >> 4) & 1) * 8;
#pragma unroll
  for (int i = 0; i < 8; ++i) {
    int k = 2 * (i & 3) + ((i >> 2) << 4);
    f.u[i] = *(const unsigned int*)(p + k);
  }
  return f.v;
}

// B-matrix fragment (32x16 = KxN): weight stored natural (N_out rows, K_in cols), W[n][k]=Wt.T[k][n].
// lane l -> col l&15; VGPR i -> K = 2*(i&3) + 16*(i>>2) + 8*(l>>4).
__device__ __forceinline__ v16bf fragB(const unsigned short* w, int n0, int K, int k0, int lane) {
  FragU f;
  const unsigned short* p = w + (size_t)(n0 + (lane & 15)) * K + k0 + ((lane >> 4) & 1) * 8;
#pragma unroll
  for (int i = 0; i < 8; ++i) f.u[i] = *(const unsigned int*)(p + 2 * (i & 3) + ((i >> 2) << 4));
  return f.v;
}

#define WMMA_BF16(A, B, C) \
  __builtin_amdgcn_wmma_f32_16x16x32_bf16(false, (A), false, (B), (short)0, (C), false, false)

// ================= prep: gather k = k_emb[q] (bf16) and yq = r * a_W[:, :C].T[q] =================
__global__ __launch_bounds__(128) void skvmn_gather(
    const int* __restrict__ q, const int* __restrict__ r,
    const float* __restrict__ kemb, const float* __restrict__ aW,
    unsigned short* __restrict__ kbf, float* __restrict__ yqo) {
  int bt = blockIdx.x, d = threadIdx.x;
  int qq = q[bt], rr = r[bt];
  kbf[(size_t)bt * DD + d] = f2bf(kemb[(size_t)qq * DD + d]);
  yqo[(size_t)bt * DD + d] = aW[(size_t)d * (NUMC + DD) + qq] * (float)rr;
}

// ================= prep: convert all weights to bf16, natural (out,in) layout =================
__global__ __launch_bounds__(256) void skvmn_cvt_weights(
    const float* __restrict__ Mk, const float* __restrict__ fW, const float* __restrict__ aW,
    const float* __restrict__ eW, const float* __restrict__ adW,
    const float* __restrict__ Wih, const float* __restrict__ Whh,
    unsigned short* __restrict__ wtab) {
  int i = blockIdx.x * blockDim.x + threadIdx.x;
  if (i >= WT_TOTAL) return;
  float v;
  if (i < WOFF_FW)       { int j = i;            int n = j >> 7, k = j & 127; v = (n < MM) ? Mk[n * DD + k] : 0.0f; }
  else if (i < WOFF_AWF) { v = fW[i - WOFF_FW]; }
  else if (i < WOFF_EW)  { int j = i - WOFF_AWF; int n = j >> 7, k = j & 127; v = aW[(size_t)n * (NUMC + DD) + NUMC + k]; }
  else if (i < WOFF_ADW) { v = eW[i - WOFF_EW]; }
  else if (i < WOFF_WIH) { v = adW[i - WOFF_ADW]; }
  else if (i < WOFF_WHH) { v = Wih[i - WOFF_WIH]; }
  else                   { v = Whh[i - WOFF_WHH]; }
  wtab[i] = f2bf(v);
}

// ================= phase A: 512-step memory scan, persistent block per 8 batch rows =================
__global__ __launch_bounds__(256) void skvmn_mem_scan(
    const unsigned short* __restrict__ kbf, const float* __restrict__ yq,
    const unsigned short* __restrict__ wtab, const float* __restrict__ Mv0,
    const float* __restrict__ f_b, const float* __restrict__ a_b,
    const float* __restrict__ e_b, const float* __restrict__ add_b,
    float* __restrict__ wall, unsigned short* __restrict__ fbf) {
  __shared__ unsigned short sMk[MPAD * DD];        // 16 KB
  __shared__ unsigned short sFW[DD * 2 * DD];      // 64 KB
  __shared__ unsigned short sAW[DD * DD];          // 32 KB
  __shared__ unsigned short sEW[DD * DD];          // 32 KB
  __shared__ unsigned short sDW[DD * DD];          // 32 KB
  __shared__ unsigned short sMv[BPB * MM * DD];    // 100 KB (recurrent state, bf16)
  __shared__ unsigned short sX[16 * 256];          // 8 KB: [read | k] A-staging; cols 0..127 reused for y
  __shared__ unsigned short sF[16 * DD];           // 4 KB
  __shared__ float sLog[BPB * MPAD];
  __shared__ float sW[BPB * MPAD];
  __shared__ float sYq[BPB * DD];
  __shared__ float sE[BPB * DD];
  __shared__ float sAdd[BPB * DD];
  __shared__ float sBias[4 * DD];

  const int tid = threadIdx.x, lane = tid & 31, wv = tid >> 5;
  const int bb0 = blockIdx.x * BPB;

  for (int i = tid; i < MPAD * DD; i += 256) sMk[i] = wtab[WOFF_MK + i];
  for (int i = tid; i < DD * 2 * DD; i += 256) sFW[i] = wtab[WOFF_FW + i];
  for (int i = tid; i < DD * DD; i += 256) {
    sAW[i] = wtab[WOFF_AWF + i]; sEW[i] = wtab[WOFF_EW + i]; sDW[i] = wtab[WOFF_ADW + i];
  }
  for (int i = tid; i < BPB * MM * DD; i += 256) sMv[i] = f2bf(Mv0[i % (MM * DD)]);
  for (int i = tid; i < DD; i += 256) {
    sBias[i] = f_b[i]; sBias[DD + i] = a_b[i]; sBias[2 * DD + i] = e_b[i]; sBias[3 * DD + i] = add_b[i];
  }
  for (int i = tid; i < 16 * 256; i += 256) sX[i] = 0;   // pad rows 8..15 stay zero forever
  for (int i = tid; i < 16 * DD; i += 256) sF[i] = 0;

  // software pipeline: prefetch step-0 inputs into registers (4 elements/thread)
  unsigned short kpre[4]; float ypre[4];
#pragma unroll
  for (int j = 0; j < 4; ++j) {
    int i = tid + j * 256, rr = i >> 7, d = i & 127;
    size_t g = ((size_t)(bb0 + rr) * LT + 0) * DD + d;
    kpre[j] = kbf[g]; ypre[j] = yq[g];
  }
  __syncthreads();

  for (int t = 0; t < LT; ++t) {
    // ---- stage 1: commit prefetched k_t -> sX[:,128:256], yq_t -> sYq
#pragma unroll
    for (int j = 0; j < 4; ++j) {
      int i = tid + j * 256, rr = i >> 7, d = i & 127;
      sX[rr * 256 + 128 + d] = kpre[j];
      sYq[i] = ypre[j];
    }
    __syncthreads();
    // issue prefetch for t+1; s_wait lands next iteration, hidden behind the WMMA stages
    if (t + 1 < LT) {
#pragma unroll
      for (int j = 0; j < 4; ++j) {
        int i = tid + j * 256, rr = i >> 7, d = i & 127;
        size_t g = ((size_t)(bb0 + rr) * LT + (t + 1)) * DD + d;
        kpre[j] = kbf[g]; ypre[j] = yq[g];
      }
    }
    // ---- stage 2: logits(16x64) = k @ Mk^T  (waves 0..3, one N-tile each)
    if (wv < 4) {
      v8f acc = {0.f, 0.f, 0.f, 0.f, 0.f, 0.f, 0.f, 0.f};
      int n0 = wv * 16;
#pragma unroll
      for (int kk = 0; kk < 4; ++kk)
        acc = WMMA_BF16(fragA(sX, 256, 128 + kk * 32, lane), fragB(sMk, n0, DD, kk * 32, lane), acc);
      int col = n0 + (lane & 15), rb = (lane >> 4) * 8;
#pragma unroll
      for (int v = 0; v < 8; ++v) { int rr = rb + v; if (rr < BPB) sLog[rr * MPAD + col] = acc[v]; }
    }
    __syncthreads();
    // ---- stage 3: softmax over 50 slots (wave 0, 4 lanes per batch row, shfl_xor butterflies)
    if (wv == 0) {
      int row = lane >> 2, sub = lane & 3;
      float mx = -3.4e38f;
      for (int m = sub; m < MM; m += 4) { float v = sLog[row * MPAD + m]; mx = (v > mx) ? v : mx; }
      { float o = __shfl_xor(mx, 1, 4); mx = (o > mx) ? o : mx; }
      { float o = __shfl_xor(mx, 2, 4); mx = (o > mx) ? o : mx; }
      float s = 0.f;
      for (int m = sub; m < MM; m += 4) {
        float e = __expf(sLog[row * MPAD + m] - mx);
        sLog[row * MPAD + m] = e; s += e;
      }
      s += __shfl_xor(s, 1, 4);
      s += __shfl_xor(s, 2, 4);
      float inv = 1.0f / s;
      for (int m = sub; m < MPAD; m += 4) {
        float w = (m < MM) ? sLog[row * MPAD + m] * inv : 0.f;
        sW[row * MPAD + m] = w;
        if (m < MM) wall[((size_t)(bb0 + row) * LT + t) * MM + m] = w;
      }
    }
    __syncthreads();
    // ---- stage 4: read = w @ Mv  -> sX[:,0:128] (bf16)
#pragma unroll
    for (int j = 0; j < 4; ++j) {
      int i = tid + j * 256, rr = i >> 7, d = i & 127;
      const unsigned short* mv = &sMv[rr * MM * DD + d];
      const float* w = &sW[rr * MPAD];
      float acc = 0.f;
      for (int m = 0; m < MM; ++m) acc += w[m] * bf2f(mv[m * DD]);
      sX[rr * 256 + d] = f2bf(acc);
    }
    __syncthreads();
    // ---- stage 5: f = tanh([read,k] @ f_W^T + f_b)  (8 waves x 1 N-tile, K=256)
    {
      v8f acc = {0.f, 0.f, 0.f, 0.f, 0.f, 0.f, 0.f, 0.f};
      int n0 = wv * 16;
#pragma unroll
      for (int kk = 0; kk < 8; ++kk)
        acc = WMMA_BF16(fragA(sX, 256, kk * 32, lane), fragB(sFW, n0, 256, kk * 32, lane), acc);
      int col = n0 + (lane & 15), rb = (lane >> 4) * 8;
#pragma unroll
      for (int v = 0; v < 8; ++v) {
        int rr = rb + v;
        if (rr < BPB) {
          unsigned short h = f2bf(tanhf(acc[v] + sBias[col]));
          sF[rr * DD + col] = h;
          fbf[((size_t)(bb0 + rr) * LT + t) * DD + col] = h;
        }
      }
    }
    __syncthreads();
    // ---- stage 6: y = f @ a_Wf^T + yq + a_b  -> sX[:,0:128] (bf16)
    {
      v8f acc = {0.f, 0.f, 0.f, 0.f, 0.f, 0.f, 0.f, 0.f};
      int n0 = wv * 16;
#pragma unroll
      for (int kk = 0; kk < 4; ++kk)
        acc = WMMA_BF16(fragA(sF, DD, kk * 32, lane), fragB(sAW, n0, DD, kk * 32, lane), acc);
      int col = n0 + (lane & 15), rb = (lane >> 4) * 8;
#pragma unroll
      for (int v = 0; v < 8; ++v) {
        int rr = rb + v;
        if (rr < BPB) sX[rr * 256 + col] = f2bf(acc[v] + sYq[rr * DD + col] + sBias[DD + col]);
      }
    }
    __syncthreads();
    // ---- stage 7: e = sigmoid(y @ e_W^T + e_b), a = tanh(y @ add_W^T + add_b)
    {
      v8f ae = {0.f, 0.f, 0.f, 0.f, 0.f, 0.f, 0.f, 0.f};
      v8f aa = {0.f, 0.f, 0.f, 0.f, 0.f, 0.f, 0.f, 0.f};
      int n0 = wv * 16;
#pragma unroll
      for (int kk = 0; kk < 4; ++kk) {
        v16bf a = fragA(sX, 256, kk * 32, lane);
        ae = WMMA_BF16(a, fragB(sEW, n0, DD, kk * 32, lane), ae);
        aa = WMMA_BF16(a, fragB(sDW, n0, DD, kk * 32, lane), aa);
      }
      int col = n0 + (lane & 15), rb = (lane >> 4) * 8;
#pragma unroll
      for (int v = 0; v < 8; ++v) {
        int rr = rb + v;
        if (rr < BPB) {
          sE[rr * DD + col]   = sigmf(ae[v] + sBias[2 * DD + col]);
          sAdd[rr * DD + col] = tanhf(aa[v] + sBias[3 * DD + col]);
        }
      }
    }
    __syncthreads();
    // ---- stage 8: Mv = Mv*(1 - w*e) + w*a
    for (int i = tid; i < BPB * MM * DD; i += 256) {
      int d = i & 127, m = (i >> 7) % MM, b = i / (MM * DD);
      float w = sW[b * MPAD + m];
      float mv = bf2f(sMv[i]);
      sMv[i] = f2bf(mv * (1.0f - w * sE[b * DD + d]) + w * sAdd[b * DD + d]);
    }
    __syncthreads();
  }
}

// ================= phase B1: iv thresholds -> 100-bit pattern code =================
__global__ __launch_bounds__(256) void skvmn_codes(
    const float* __restrict__ wall, unsigned long long* __restrict__ codes) {
  int bt = blockIdx.x * blockDim.x + threadIdx.x;
  if (bt >= BSZ * LT) return;
  const float* w = wall + (size_t)bt * MM;
  unsigned long long lo = 0ull, hi = 0ull;
  for (int m = 0; m < MM; ++m) {
    float x = w[m];
    float tw = fmaxf(fminf((x - 0.075f) / (0.088f - 0.075f), (1.0f - x) / (1.0f - 0.088f)), 0.0f);
    unsigned long long lvl = (tw >= 0.6f) ? 2ull : ((tw >= 0.1f) ? 1ull : 0ull);
    if (m < 32) lo |= lvl << (2 * m); else hi |= lvl << (2 * (m - 32));
  }
  codes[2 * bt] = lo; codes[2 * bt + 1] = hi;
}

// ================= phase B2: prev[i] = largest j<i with identical pattern (else -1) =================
__global__ __launch_bounds__(512) void skvmn_match(
    const unsigned long long* __restrict__ codes, int* __restrict__ prevM) {
  __shared__ unsigned long long slo[LT], shi[LT];
  int b = blockIdx.x, i = threadIdx.x;
  slo[i] = codes[2 * ((size_t)b * LT + i)];
  shi[i] = codes[2 * ((size_t)b * LT + i) + 1];
  __syncthreads();
  unsigned long long l = slo[i], h = shi[i];
  int best = -1;
  for (int j = i - 1; j >= 0; --j) if (slo[j] == l && shi[j] == h) { best = j; break; }
  prevM[(size_t)b * LT + i] = best;
}

// ================= phase C: skip-LSTM scan + output head =================
__global__ __launch_bounds__(256) void skvmn_lstm(
    const unsigned short* __restrict__ fbf, const unsigned short* __restrict__ wtab,
    const int* __restrict__ prevM,
    const float* __restrict__ bih, const float* __restrict__ bhh,
    const float* __restrict__ pW, const float* __restrict__ pb,
    float* __restrict__ Hh, float* __restrict__ Ch, float* __restrict__ out) {
  __shared__ unsigned short sWih[4 * DD * DD];   // 128 KB
  __shared__ unsigned short sWhh[4 * DD * DD];   // 128 KB
  __shared__ unsigned short sAf[16 * DD];        // f_t A-staging
  __shared__ unsigned short sAh[16 * DD];        // h_in A-staging
  __shared__ float sG[BPB * 4 * DD];             // gates
  __shared__ float sH[BPB * DD];
  __shared__ float sC[BPB * DD];
  __shared__ float sB[4 * DD];
  __shared__ float sPW[DD];

  const int tid = threadIdx.x, lane = tid & 31, wv = tid >> 5;
  const int bb0 = blockIdx.x * BPB;
  const float pbias = pb[0];

  for (int i = tid; i < 4 * DD * DD; i += 256) { sWih[i] = wtab[WOFF_WIH + i]; sWhh[i] = wtab[WOFF_WHH + i]; }
  for (int i = tid; i < 4 * DD; i += 256) sB[i] = bih[i] + bhh[i];
  for (int i = tid; i < DD; i += 256) sPW[i] = pW[i];
  for (int i = tid; i < 16 * DD; i += 256) { sAf[i] = 0; sAh[i] = 0; }
  for (int i = tid; i < BPB * DD; i += 256) { sH[i] = 0.f; sC[i] = 0.f; }

  // software pipeline: prefetch step-0 f_t and prev indices
  unsigned short fpre[4]; int pmpre[4];
#pragma unroll
  for (int j = 0; j < 4; ++j) {
    int i = tid + j * 256, rr = i >> 7, d = i & 127, bg = bb0 + rr;
    fpre[j]  = fbf[((size_t)bg * LT + 0) * DD + d];
    pmpre[j] = prevM[(size_t)bg * LT + 0];
  }
  __syncthreads();

  for (int t = 0; t < LT; ++t) {
    // ---- stage 1: select h_in/c_in (skip read, agent-scope loads bypass stale L0) + commit f_t
#pragma unroll
    for (int j = 0; j < 4; ++j) {
      int i = tid + j * 256, rr = i >> 7, d = i & 127, bg = bb0 + rr;
      int pm = pmpre[j];
      float hin, cin;
      if (pm >= 0) {
        size_t g = ((size_t)pm * BSZ + bg) * DD + d;
        hin = __hip_atomic_load(&Hh[g], __ATOMIC_RELAXED, __HIP_MEMORY_SCOPE_AGENT);
        cin = __hip_atomic_load(&Ch[g], __ATOMIC_RELAXED, __HIP_MEMORY_SCOPE_AGENT);
      } else { hin = sH[i]; cin = sC[i]; }
      sAh[rr * DD + d] = f2bf(hin);
      sC[i] = cin;
      sAf[rr * DD + d] = fpre[j];
    }
    __syncthreads();
    // prefetch step t+1 inputs (latency hidden behind the gate GEMM)
    if (t + 1 < LT) {
#pragma unroll
      for (int j = 0; j < 4; ++j) {
        int i = tid + j * 256, rr = i >> 7, d = i & 127, bg = bb0 + rr;
        fpre[j]  = fbf[((size_t)bg * LT + (t + 1)) * DD + d];
        pmpre[j] = prevM[(size_t)bg * LT + (t + 1)];
      }
    }
    // ---- stage 2: gates(16x512) = f @ Wih^T + h_in @ Whh^T + b  (8 waves x 4 N-tiles)
    for (int tile = wv; tile < 32; tile += 8) {
      v8f acc = {0.f, 0.f, 0.f, 0.f, 0.f, 0.f, 0.f, 0.f};
      int n0 = tile * 16;
#pragma unroll
      for (int kk = 0; kk < 4; ++kk)
        acc = WMMA_BF16(fragA(sAf, DD, kk * 32, lane), fragB(sWih, n0, DD, kk * 32, lane), acc);
#pragma unroll
      for (int kk = 0; kk < 4; ++kk)
        acc = WMMA_BF16(fragA(sAh, DD, kk * 32, lane), fragB(sWhh, n0, DD, kk * 32, lane), acc);
      int col = n0 + (lane & 15), rb = (lane >> 4) * 8;
#pragma unroll
      for (int v = 0; v < 8; ++v) { int rr = rb + v; if (rr < BPB) sG[rr * 512 + col] = acc[v] + sB[col]; }
    }
    __syncthreads();
    // ---- stage 3: LSTM cell update; history stored agent-scope (write-through past L0)
#pragma unroll
    for (int j = 0; j < 4; ++j) {
      int i = tid + j * 256, rr = i >> 7, d = i & 127, bg = bb0 + rr;
      float ig = sG[rr * 512 + d],       fg = sG[rr * 512 + 128 + d];
      float gg = sG[rr * 512 + 256 + d], og = sG[rr * 512 + 384 + d];
      float cn = sigmf(fg) * sC[i] + sigmf(ig) * tanhf(gg);
      float hn = sigmf(og) * tanhf(cn);
      sC[i] = cn; sH[i] = hn;
      size_t g = ((size_t)t * BSZ + bg) * DD + d;
      __hip_atomic_store(&Hh[g], hn, __ATOMIC_RELAXED, __HIP_MEMORY_SCOPE_AGENT);
      __hip_atomic_store(&Ch[g], cn, __ATOMIC_RELAXED, __HIP_MEMORY_SCOPE_AGENT);
    }
    __threadfence();   // drain STOREcnt: history must be globally visible before readers pass barrier
    __syncthreads();
    // ---- stage 4: p = sigmoid(h . p_W + p_b)
    if (tid < BPB) {
      float acc = pbias;
      for (int d = 0; d < DD; ++d) acc += sH[tid * DD + d] * sPW[d];
      out[(size_t)(bb0 + tid) * LT + t] = sigmf(acc);
    }
    __syncthreads();
  }
}

extern "C" void kernel_launch(void* const* d_in, const int* in_sizes, int n_in,
                              void* d_out, int out_size, void* d_ws, size_t ws_size,
                              hipStream_t stream) {
  const int*   q     = (const int*)d_in[0];
  const int*   r     = (const int*)d_in[1];
  const float* k_emb = (const float*)d_in[2];
  const float* Mk    = (const float*)d_in[3];
  const float* Mv0   = (const float*)d_in[4];
  const float* f_W   = (const float*)d_in[5];
  const float* f_b   = (const float*)d_in[6];
  const float* a_W   = (const float*)d_in[7];
  const float* a_b   = (const float*)d_in[8];
  const float* e_W   = (const float*)d_in[9];
  const float* e_b   = (const float*)d_in[10];
  const float* add_W = (const float*)d_in[11];
  const float* add_b = (const float*)d_in[12];
  const float* Wih   = (const float*)d_in[13];
  const float* Whh   = (const float*)d_in[14];
  const float* bih   = (const float*)d_in[15];
  const float* bhh   = (const float*)d_in[16];
  const float* p_W   = (const float*)d_in[17];
  const float* p_b   = (const float*)d_in[18];
  float* out = (float*)d_out;

  char* ws = (char*)d_ws;
  unsigned short*     kbf   = (unsigned short*)(ws + WS_KBF);
  float*              yq    = (float*)(ws + WS_YQ);
  unsigned short*     fbf   = (unsigned short*)(ws + WS_FBF);
  float*              wall  = (float*)(ws + WS_WALL);
  unsigned long long* codes = (unsigned long long*)(ws + WS_CODE);
  int*                prevM = (int*)(ws + WS_PREV);
  float*              Hh    = (float*)(ws + WS_HH);
  float*              Ch    = (float*)(ws + WS_CH);
  unsigned short*     wtab  = (unsigned short*)(ws + WS_WTAB);

  skvmn_gather<<<BSZ * LT, 128, 0, stream>>>(q, r, k_emb, a_W, kbf, yq);
  skvmn_cvt_weights<<<(WT_TOTAL + 255) / 256, 256, 0, stream>>>(Mk, f_W, a_W, e_W, add_W, Wih, Whh, wtab);
  skvmn_mem_scan<<<BSZ / BPB, 256, 0, stream>>>(kbf, yq, wtab, Mv0, f_b, a_b, e_b, add_b, wall, fbf);
  skvmn_codes<<<(BSZ * LT + 255) / 256, 256, 0, stream>>>(wall, codes);
  skvmn_match<<<BSZ, LT, 0, stream>>>(codes, prevM);
  skvmn_lstm<<<BSZ / BPB, 256, 0, stream>>>(fbf, wtab, prevM, bih, bhh, p_W, p_b, Hh, Ch, out);
}
